// Separate_Non_Local_40905268527206
// MI455X (gfx1250) — compile-verified, matrix operand order checked by reading, not run on hardware
//
#include <hip/hip_runtime.h>

// ---------------- problem constants ----------------
#define Bsz 4
#define Tn  7
#define Cc  64
#define C2  32
#define HWn 4096
#define THW (Tn*HWn)            // 28672
#define Khw (C2*Tn)             // 224
#define PROJ_SZ ((size_t)Bsz*C2*Tn*HWn)   // 3,670,016 elements

// ---------------- WMMA helpers ----------------
typedef __attribute__((ext_vector_type(16))) _Float16 v16h;
typedef __attribute__((ext_vector_type(8)))  _Float16 v8h;
typedef __attribute__((ext_vector_type(8)))  float    v8f;

// Per-lane operand fetch for v_wmma_f32_16x16x32_f16.
// A: rowptr = &A[row(lane)][0] (row-major, K contiguous)
// B: rowptr = &B[col(lane)][0] (col-major, K contiguous)
// halves = {k0..k0+7, k0+16..k0+23}, k0 = kbase + ((lane&16)?8:0)
__device__ __forceinline__ v16h ld_ab(const _Float16* rowptr, int kbase, int lane) {
  int k0 = kbase + ((lane & 16) ? 8 : 0);
  v16h r;
  ((v8h*)&r)[0] = *(const v8h*)(rowptr + k0);
  ((v8h*)&r)[1] = *(const v8h*)(rowptr + k0 + 16);
  return r;
}

__device__ __forceinline__ v8f wmma16(v16h a, v16h b, v8f c) {
  return __builtin_amdgcn_wmma_f32_16x16x32_f16(false, a, false, b, (short)0, c, false, false);
}

struct WB9 { const float* w[9]; const float* b[9]; };
struct ZWB { const float* w[3]; const float* b[3]; };

// ============================================================
// K1: all nine 1x1-conv projections, one (b,t) x 128-col strip per block.
//   D[288,4096] = Wall[288,64] * X[64,4096] (+bias), via WMMA 16x16x32.
//   hw_theta / hw_phi are written transposed [B][4096][224];
//   the other 7 projections go to pg slots [B][32][T][HW].
// ============================================================
__global__ void k_proj(const float* __restrict__ x, WB9 wb,
                       _Float16* __restrict__ thT, _Float16* __restrict__ phT,
                       _Float16* __restrict__ pg) {
  __shared__ _Float16 W16[9*2048];   // [9][32][64] f16
  __shared__ float    BL[288];
  __shared__ _Float16 Xl[128*64];    // [j][c] f16 (K contiguous per column)
  int tid = threadIdx.x;
  int bt = blockIdx.y; int b = bt / Tn, t = bt % Tn;
  int hw0 = blockIdx.x * 128;

#pragma unroll
  for (int wi = 0; wi < 9; ++wi) {
    const float* wp = wb.w[wi];
    for (int j = tid; j < 2048; j += 256) W16[wi*2048 + j] = (_Float16)wp[j];
    if (tid < 32) BL[wi*32 + tid] = wb.b[wi][tid];
  }
  const float* xp = x + ((size_t)bt * 64) * 4096 + hw0;
  for (int i = tid; i < 64*128; i += 256) {
    int c = i >> 7, j = i & 127;
    Xl[j*64 + c] = (_Float16)xp[(size_t)c*4096 + j];
  }
  __syncthreads();

  int wave = tid >> 5, lane = tid & 31, l15 = lane & 15;
  for (int job = wave; job < 144; job += 8) {        // 18 m-tiles x 8 n-tiles
    int mt = job >> 3, nt = job & 7;
    v8f acc = {};
    const _Float16* arow = W16 + (mt*16 + l15)*64;
    const _Float16* brow = Xl  + (nt*16 + l15)*64;
    acc = wmma16(ld_ab(arow, 0,  lane), ld_ab(brow, 0,  lane), acc);
    acc = wmma16(ld_ab(arow, 32, lane), ld_ab(brow, 32, lane), acc);

    int i9 = mt >> 1, ob = (mt & 1) * 16;
    int n = l15, moff = (lane & 16) ? 8 : 0;
    if (i9 < 2) {                                    // hw theta/phi -> transposed
      _Float16* dst = i9 ? phT : thT;
#pragma unroll
      for (int v = 0; v < 8; ++v) {
        int ml = v + moff;  int o = ob + ml;
        float val = acc[v] + BL[mt*16 + ml];
        dst[((size_t)(b*4096) + hw0 + nt*16 + n)*224 + o*7 + t] = (_Float16)val;
      }
    } else {
      _Float16* dst = pg + (size_t)(i9 - 2)*PROJ_SZ;
#pragma unroll
      for (int v = 0; v < 8; ++v) {
        int ml = v + moff;  int o = ob + ml;
        float val = acc[v] + BL[mt*16 + ml];
        dst[(((size_t)(b*32 + o))*7 + t)*4096 + hw0 + nt*16 + n] = (_Float16)val;
      }
    }
  }
}

// ============================================================
// K2: HW-branch flash attention. One block = one batch x 16 p-rows.
//   S = Q^T K (d=224), online softmax over 16 q-chunks of 256,
//   O += P * G^T accumulated in WMMA accs (14 c-tiles across 7 waves).
//   Output y written transposed [B][T][HW][32].
// ============================================================
__global__ void k_attn_hw(const _Float16* __restrict__ thT,
                          const _Float16* __restrict__ phT,
                          const _Float16* __restrict__ g,      // [B][224][4096]
                          _Float16* __restrict__ yhwT) {       // [B][T][HW][32]
  __shared__ _Float16 Ql[16*224];
  __shared__ float    Sl[16*256];
  __shared__ _Float16 Pl[16*256];
  __shared__ float    red[256];
  __shared__ float    rmax[16], rsum[16], scal[16];

  int tid = threadIdx.x, wave = tid >> 5, lane = tid & 31, l15 = lane & 15;
  int b = blockIdx.y;
  int pbase = blockIdx.x * 16;

  for (int i = tid; i < 16*224; i += 256)
    Ql[i] = thT[((size_t)(b*4096 + pbase + i/224))*224 + (i%224)];
  if (tid < 16) { rmax[tid] = -1e30f; rsum[tid] = 0.f; }
  __syncthreads();

  v8f o0 = {}, o1 = {};
  int ct0 = 2*wave, ct1 = 2*wave + 1;   // waves 0..6 own two c-tiles (14 total)

  for (int ch = 0; ch < 16; ++ch) {
    int q0 = ch * 256;
    // ---- S tiles (each wave computes 2 of 16 q-subtiles) ----
#pragma unroll
    for (int s = 0; s < 2; ++s) {
      int qt = 2*wave + s;
      v8f sa = {};
      const _Float16* arow = Ql + l15*224;
      const _Float16* brow = phT + ((size_t)(b*4096 + q0 + qt*16 + l15))*224;
#pragma unroll
      for (int kb = 0; kb < 224; kb += 32)
        sa = wmma16(ld_ab(arow, kb, lane), ld_ab(brow, kb, lane), sa);
      int moff = (lane & 16) ? 8 : 0;
#pragma unroll
      for (int v = 0; v < 8; ++v)
        Sl[(v + moff)*256 + qt*16 + l15] = sa[v];
    }
    __syncthreads();
    // ---- online softmax: chunk max ----
    {
      int r = tid >> 4, seg = tid & 15;
      float lm = -1e30f;
#pragma unroll
      for (int j = 0; j < 16; ++j) lm = fmaxf(lm, Sl[r*256 + seg*16 + j]);
      red[tid] = lm;
    }
    __syncthreads();
    if (tid < 16) {
      float m = rmax[tid];
#pragma unroll
      for (int j = 0; j < 16; ++j) m = fmaxf(m, red[tid*16 + j]);
      scal[tid] = __expf(rmax[tid] - m);
      rmax[tid] = m;
    }
    __syncthreads();
    // ---- exp + partial sums, P to LDS (f16) ----
    {
      int r = tid >> 4, seg = tid & 15;
      float m = rmax[r], ls = 0.f;
#pragma unroll
      for (int j = 0; j < 16; ++j) {
        float e = __expf(Sl[r*256 + seg*16 + j] - m);
        Pl[r*256 + seg*16 + j] = (_Float16)e;
        ls += e;
      }
      red[tid] = ls;
    }
    __syncthreads();
    if (tid < 16) {
      float ssum = 0.f;
#pragma unroll
      for (int j = 0; j < 16; ++j) ssum += red[tid*16 + j];
      rsum[tid] = rsum[tid]*scal[tid] + ssum;
    }
    __syncthreads();
    // ---- rescale accs, accumulate O += P * G^T ----
    if (ct0 < 14) {
      int moff = (lane & 16) ? 8 : 0;
#pragma unroll
      for (int v = 0; v < 8; ++v) { float sc = scal[v + moff]; o0[v] *= sc; o1[v] *= sc; }
      const _Float16* arow  = Pl + l15*256;
      const _Float16* brow0 = g + ((size_t)(b*224 + ct0*16 + l15))*4096 + q0;
      const _Float16* brow1 = g + ((size_t)(b*224 + ct1*16 + l15))*4096 + q0;
#pragma unroll
      for (int kb = 0; kb < 256; kb += 32) {
        v16h pa = ld_ab(arow, kb, lane);
        o0 = wmma16(pa, ld_ab(brow0, kb, lane), o0);
        o1 = wmma16(pa, ld_ab(brow1, kb, lane), o1);
      }
    }
    __syncthreads();
  }
  // ---- normalize and write y (transposed layout) ----
  if (ct0 < 14) {
    int moff = (lane & 16) ? 8 : 0;
#pragma unroll
    for (int s = 0; s < 2; ++s) {
      int ct = s ? ct1 : ct0;
#pragma unroll
      for (int v = 0; v < 8; ++v) {
        int m = v + moff;
        int p = pbase + m;
        int c = ct*16 + l15;
        int o = c / 7, tt = c % 7;
        float val = (s ? o1[v] : o0[v]) / rsum[m];
        yhwT[(((size_t)(b*7 + tt))*4096 + p)*32 + o] = (_Float16)val;
      }
    }
  }
}

// ============================================================
// K3a: C-branch scores f = softmax(th * ph^T), [32x32] per batch, K=28672 WMMA.
// ============================================================
__global__ void k_attn_c_scores(const _Float16* __restrict__ th,
                                const _Float16* __restrict__ ph,
                                float* __restrict__ fc) {
  __shared__ float Sc[32*32];
  int tid = threadIdx.x, wave = tid >> 5, lane = tid & 31, l15 = lane & 15;
  int b = blockIdx.x;
  if (wave < 4) {
    int mt = wave >> 1, nt = wave & 1;
    v8f acc = {};
    const _Float16* arow = th + ((size_t)(b*32 + mt*16 + l15))*28672;
    const _Float16* brow = ph + ((size_t)(b*32 + nt*16 + l15))*28672;
    for (int kb = 0; kb < 28672; kb += 32)
      acc = wmma16(ld_ab(arow, kb, lane), ld_ab(brow, kb, lane), acc);
    int moff = (lane & 16) ? 8 : 0;
#pragma unroll
    for (int v = 0; v < 8; ++v)
      Sc[(mt*16 + v + moff)*32 + nt*16 + l15] = acc[v];
  }
  __syncthreads();
  if (tid < 32) {
    float m = -1e30f;
#pragma unroll
    for (int j = 0; j < 32; ++j) m = fmaxf(m, Sc[tid*32 + j]);
    float e[32], s = 0.f;
#pragma unroll
    for (int j = 0; j < 32; ++j) { e[j] = __expf(Sc[tid*32 + j] - m); s += e[j]; }
    float inv = 1.f / s;
#pragma unroll
    for (int j = 0; j < 32; ++j) fc[(b*32 + tid)*32 + j] = e[j]*inv;
  }
}

// K3b: y_c = f * g  (32 rows per spatial column; 1024 FMAs/thread, f from LDS)
__global__ void k_attn_c_apply(const float* __restrict__ fc,
                               const _Float16* __restrict__ gc,
                               _Float16* __restrict__ ycT) {
  __shared__ float fl[1024];
  int tid = threadIdx.x;
  size_t idx = (size_t)blockIdx.x*256 + tid;
  int b = blockIdx.x / 112;                 // 28672/256 = 112 blocks per batch
  int n = (int)(idx - (size_t)b*28672);
  for (int i = tid; i < 1024; i += 256) fl[i] = fc[b*1024 + i];
  __syncthreads();
  float acc[32];
#pragma unroll
  for (int i = 0; i < 32; ++i) acc[i] = 0.f;
  for (int j = 0; j < 32; ++j) {
    float gv = (float)gc[((size_t)(b*32 + j))*28672 + n];
#pragma unroll
    for (int i = 0; i < 32; ++i) acc[i] += fl[i*32 + j]*gv;
  }
  int t = n >> 12, hw = n & 4095;
  size_t obase = (((size_t)(b*7 + t))*4096 + hw)*32;
#pragma unroll
  for (int i = 0; i < 32; ++i) ycT[obase + i] = (_Float16)acc[i];
}

// ============================================================
// K4a: T-branch scores f = softmax(7x7) per batch, reduce over d=131072.
// ============================================================
__global__ void k_attn_t_scores(const _Float16* __restrict__ th,
                                const _Float16* __restrict__ ph,
                                float* __restrict__ ft) {
  __shared__ float red[256];
  __shared__ float S49[49];
  int tid = threadIdx.x, b = blockIdx.x;
  float s[49];
#pragma unroll
  for (int i = 0; i < 49; ++i) s[i] = 0.f;
  for (int d = tid; d < 131072; d += 256) {
    int o = d >> 12, hw = d & 4095;
    float tv[7], pv[7];
#pragma unroll
    for (int t = 0; t < 7; ++t) {
      size_t a = (((size_t)(b*32 + o))*7 + t)*4096 + hw;
      tv[t] = (float)th[a]; pv[t] = (float)ph[a];
    }
#pragma unroll
    for (int t = 0; t < 7; ++t)
#pragma unroll
      for (int u = 0; u < 7; ++u) s[t*7 + u] += tv[t]*pv[u];
  }
  for (int k = 0; k < 49; ++k) {
    red[tid] = s[k]; __syncthreads();
    for (int off = 128; off > 0; off >>= 1) {
      if (tid < off) red[tid] += red[tid + off];
      __syncthreads();
    }
    if (tid == 0) S49[k] = red[0];
    __syncthreads();
  }
  if (tid < 7) {
    float m = -1e30f;
#pragma unroll
    for (int u = 0; u < 7; ++u) m = fmaxf(m, S49[tid*7 + u]);
    float e[7], ssum = 0.f;
#pragma unroll
    for (int u = 0; u < 7; ++u) { e[u] = __expf(S49[tid*7 + u] - m); ssum += e[u]; }
#pragma unroll
    for (int u = 0; u < 7; ++u) ft[b*49 + tid*7 + u] = e[u]/ssum;
  }
}

// K4b: y_t = f * g (7x7 mixing of frames per (b,o,hw))
__global__ void k_attn_t_apply(const float* __restrict__ ft,
                               const _Float16* __restrict__ gt,
                               _Float16* __restrict__ ytT) {
  __shared__ float fl[49];
  int tid = threadIdx.x;
  size_t idx = (size_t)blockIdx.x*256 + tid;
  int b = (int)(idx >> 17);
  int rem = (int)(idx & 131071);
  int o = rem >> 12, hw = rem & 4095;
  if (tid < 49) fl[tid] = ft[b*49 + tid];
  __syncthreads();
  float gv[7];
#pragma unroll
  for (int u = 0; u < 7; ++u)
    gv[u] = (float)gt[(((size_t)(b*32 + o))*7 + u)*4096 + hw];
#pragma unroll
  for (int t = 0; t < 7; ++t) {
    float y = 0.f;
#pragma unroll
    for (int u = 0; u < 7; ++u) y += fl[t*7 + u]*gv[u];
    ytT[(((size_t)(b*7 + t))*4096 + hw)*32 + o] = (_Float16)y;
  }
}

// ============================================================
// K5: fused z-convs (3x [64,32] GEMM, one WMMA K-step each) + bias + residual.
//   out[b,t,c,hw] = x + sum_p (zW_p * y_p + zb_p)
// ============================================================
__global__ void k_final(const float* __restrict__ x,
                        const _Float16* __restrict__ yhwT,
                        const _Float16* __restrict__ ycT,
                        const _Float16* __restrict__ ytT,
                        ZWB z, float* __restrict__ out) {
  __shared__ _Float16 ZW[3*2048];   // [3][64][32] f16
  __shared__ float    ZB[64];
  int tid = threadIdx.x, wave = tid >> 5, lane = tid & 31, l15 = lane & 15;
  int bt = blockIdx.x;
  int hw0 = blockIdx.y * 128;
#pragma unroll
  for (int p = 0; p < 3; ++p) {
    const float* wp = z.w[p];
    for (int j = tid; j < 2048; j += 128) ZW[p*2048 + j] = (_Float16)wp[j];
  }
  if (tid < 64) ZB[tid] = z.b[0][tid] + z.b[1][tid] + z.b[2][tid];
  __syncthreads();

  const _Float16* ys[3] = { yhwT, ycT, ytT };
  int mt = wave;                    // 4 waves -> 4 c-tiles (64 channels)
  for (int nt = 0; nt < 8; ++nt) {
    v8f acc = {};
#pragma unroll
    for (int p = 0; p < 3; ++p) {
      const _Float16* arow = ZW + (p*2048) + (mt*16 + l15)*32;
      const _Float16* brow = ys[p] + ((size_t)bt*4096 + hw0 + nt*16 + l15)*32;
      acc = wmma16(ld_ab(arow, 0, lane), ld_ab(brow, 0, lane), acc);
    }
    int moff = (lane & 16) ? 8 : 0;
#pragma unroll
    for (int v = 0; v < 8; ++v) {
      int c = mt*16 + v + moff;
      size_t a = ((size_t)bt*64 + c)*4096 + hw0 + nt*16 + l15;
      out[a] = acc[v] + ZB[c] + x[a];
    }
  }
}

// ============================================================
extern "C" void kernel_launch(void* const* d_in, const int* in_sizes, int n_in,
                              void* d_out, int out_size, void* d_ws, size_t ws_size,
                              hipStream_t stream) {
  (void)in_sizes; (void)n_in; (void)out_size; (void)ws_size;
  const float* x = (const float*)d_in[0];

  WB9 wb;
  const int wi_idx[9] = {1,3,5, 9,11,13, 17,19,21};
  const int bi_idx[9] = {2,4,6, 10,12,14, 18,20,22};
  for (int i = 0; i < 9; ++i) {
    wb.w[i] = (const float*)d_in[wi_idx[i]];
    wb.b[i] = (const float*)d_in[bi_idx[i]];
  }
  ZWB zwb;
  zwb.w[0] = (const float*)d_in[7];  zwb.b[0] = (const float*)d_in[8];
  zwb.w[1] = (const float*)d_in[15]; zwb.b[1] = (const float*)d_in[16];
  zwb.w[2] = (const float*)d_in[23]; zwb.b[2] = (const float*)d_in[24];

  // workspace carve (fp16 buffers, ~88 MB total)
  _Float16* thT = (_Float16*)d_ws;              // [B][4096][224]
  _Float16* phT = thT + PROJ_SZ;                // [B][4096][224]
  _Float16* pg  = phT + PROJ_SZ;                // 7 slots: hw_g, c_{t,p,g}, t_{t,p,g}
  _Float16* yhw = pg  + 7*PROJ_SZ;              // [B][T][HW][32]
  _Float16* yc  = yhw + PROJ_SZ;
  _Float16* yt  = yc  + PROJ_SZ;
  float*    fc  = (float*)(yt + PROJ_SZ);       // [B][32][32]
  float*    ft  = fc + Bsz*32*32;               // [B][7][7]

  const _Float16* g_hw = pg + 0*PROJ_SZ;
  const _Float16* th_c = pg + 1*PROJ_SZ;
  const _Float16* ph_c = pg + 2*PROJ_SZ;
  const _Float16* g_c  = pg + 3*PROJ_SZ;
  const _Float16* th_t = pg + 4*PROJ_SZ;
  const _Float16* ph_t = pg + 5*PROJ_SZ;
  const _Float16* g_t  = pg + 6*PROJ_SZ;

  k_proj<<<dim3(32, Bsz*Tn), 256, 0, stream>>>(x, wb, thT, phT, pg);
  k_attn_hw<<<dim3(256, Bsz), 256, 0, stream>>>(thT, phT, g_hw, yhw);
  k_attn_c_scores<<<Bsz, 256, 0, stream>>>(th_c, ph_c, fc);
  k_attn_c_apply<<<(Bsz*THW)/256, 256, 0, stream>>>(fc, g_c, yc);
  k_attn_t_scores<<<Bsz, 256, 0, stream>>>(th_t, ph_t, ft);
  k_attn_t_apply<<<(Bsz*C2*HWn)/256, 256, 0, stream>>>(ft, g_t, yt);
  k_final<<<dim3(Bsz*Tn, 32), 128, 0, stream>>>(x, yhw, yc, yt, zwb, (float*)d_out);
}
